// Block_77876347011332
// MI455X (gfx1250) — matrix operand order repaired and checked by reading, not compiled, not run on hardware
//
#include <hip/hip_runtime.h>
#include <hip/hip_bf16.h>

typedef __attribute__((ext_vector_type(16))) _Float16 v16h;
typedef __attribute__((ext_vector_type(8)))  float    v8f;

union Frag { uint4 u[2]; v16h h; };
union H8   { uint4 u; _Float16 e[8]; };

#define WMMA_F16(A, B, C) \
  __builtin_amdgcn_wmma_f32_16x16x32_f16(false, (A), false, (B), (short)0, (C), false, false)

// Low 32 bits of a generic pointer to LDS == wave-relative LDS byte address
// (ISA 10.2, LDS aperture).
__device__ __forceinline__ unsigned lds_addr32(const void* p) {
  return (unsigned)(uintptr_t)p;
}

// Async global -> LDS copy, 16 bytes per lane (ASYNCcnt-tracked, no VGPR data).
__device__ __forceinline__ void async_b128(unsigned lds_dst, const void* gsrc) {
  asm volatile("global_load_async_to_lds_b128 %0, %1, off"
               :: "v"(lds_dst), "v"(gsrc) : "memory");
}

__device__ __forceinline__ void wait_async0() {
#if __has_builtin(__builtin_amdgcn_s_wait_asynccnt)
  __builtin_amdgcn_s_wait_asynccnt(0);
#else
  asm volatile("s_wait_asynccnt 0" ::: "memory");
#endif
}

// ---------------------------------------------------------------------------
// LayerNorm: one block per row of 1024, 256 threads, f32 in -> f16 out
// ---------------------------------------------------------------------------
__global__ __launch_bounds__(256) void ln_kernel(const float* __restrict__ x,
                                                 const float* __restrict__ g,
                                                 const float* __restrict__ b,
                                                 _Float16* __restrict__ out) {
  const int row = blockIdx.x;
  const int t = threadIdx.x;
  const float* xr = x + (size_t)row * 1024;
  float4 v = *(const float4*)(xr + t * 4);
  float s  = v.x + v.y + v.z + v.w;
  float sq = v.x * v.x + v.y * v.y + v.z * v.z + v.w * v.w;
  #pragma unroll
  for (int off = 16; off; off >>= 1) {
    s  += __shfl_xor(s,  off, 32);
    sq += __shfl_xor(sq, off, 32);
  }
  __shared__ float rs[8], rq[8];
  if ((t & 31) == 0) { rs[t >> 5] = s; rq[t >> 5] = sq; }
  __syncthreads();
  float S = 0.f, Q = 0.f;
  #pragma unroll
  for (int i = 0; i < 8; i++) { S += rs[i]; Q += rq[i]; }
  const float mean = S * (1.0f / 1024.0f);
  const float var  = Q * (1.0f / 1024.0f) - mean * mean;
  const float rstd = rsqrtf(var + 1e-5f);
  const float vv[4] = {v.x, v.y, v.z, v.w};
  #pragma unroll
  for (int j = 0; j < 4; j++) {
    int c = t * 4 + j;
    out[(size_t)row * 1024 + c] = (_Float16)((vv[j] - mean) * rstd * g[c] + b[c]);
  }
}

// ---------------------------------------------------------------------------
// Tiled WMMA GEMM: D[M,N] = act(scale * A[M,K](f16) @ W[K,N](f32->f16) + bias) + res
// Block tile 128x128, BK=32, 256 threads = 8 waves, each wave 32x64 (2x4 WMMA tiles)
// A tile staged via async global->LDS copies (ASYNCcnt); W tile converted
// f32->f16 through VGPRs (needs conversion, cannot be async).
// ---------------------------------------------------------------------------
template <bool HAS_BIAS, bool GELU_, bool RES, bool OUT16>
__global__ __launch_bounds__(256) void gemm_f16(const _Float16* __restrict__ A,
                                                const float* __restrict__ W,
                                                const float* __restrict__ bias,
                                                const float* __restrict__ res,
                                                float* __restrict__ outF,
                                                _Float16* __restrict__ outH,
                                                int M, int N, int K, float scale) {
  __shared__ _Float16 sA[128 * 32];   // [row][k]
  __shared__ _Float16 sB[128 * 32];   // [n][k]  (K-transposed)

  const int t = threadIdx.x;
  const int lane = t & 31, w = t >> 5;
  const int lm = lane & 15, hi = lane >> 4;
  const int wm = w & 3, wn = w >> 2;
  const int rowBase = blockIdx.y * 128;
  const int colBase = blockIdx.x * 128;

  v8f acc[2][4];
  #pragma unroll
  for (int mt = 0; mt < 2; mt++)
    #pragma unroll
    for (int nt = 0; nt < 4; nt++) acc[mt][nt] = (v8f){};

  const int la_row = t >> 1, la_col = (t & 1) * 16;
  const int lb_row = t >> 3, lb_col = (t & 7) * 16;
  const _Float16* Ag = A + (size_t)(rowBase + la_row) * K + la_col;
  const float*    Wg = W + (size_t)lb_row * N + colBase + lb_col;
  const unsigned  sA_dst = lds_addr32(&sA[la_row * 32 + la_col]);

  for (int k0 = 0; k0 < K; k0 += 32) {
    // stage A tile: async global -> LDS (pure copy, no VGPR round-trip)
    async_b128(sA_dst,      Ag + k0);
    async_b128(sA_dst + 16, Ag + k0 + 8);
    // stage W tile (f32 -> f16, K-transposed scatter)
    const float* wp = Wg + (size_t)k0 * N;
    float4 f0 = *(const float4*)(wp);
    float4 f1 = *(const float4*)(wp + 4);
    float4 f2 = *(const float4*)(wp + 8);
    float4 f3 = *(const float4*)(wp + 12);
    const float tmp[16] = {f0.x, f0.y, f0.z, f0.w, f1.x, f1.y, f1.z, f1.w,
                           f2.x, f2.y, f2.z, f2.w, f3.x, f3.y, f3.z, f3.w};
    #pragma unroll
    for (int j = 0; j < 16; j++) sB[(lb_col + j) * 32 + lb_row] = (_Float16)tmp[j];
    if (k0 + 32 < K) {
      __builtin_prefetch(Ag + k0 + 32, 0, 1);
      __builtin_prefetch(Wg + (size_t)(k0 + 32) * N, 0, 1);
    }
    wait_async0();      // barrier does not cover ASYNCcnt
    __syncthreads();

    Frag aF[2], bF[4];
    #pragma unroll
    for (int mt = 0; mt < 2; mt++) {
      int r0 = (wm * 32 + mt * 16 + lm) * 32;
      aF[mt].u[0] = *(const uint4*)&sA[r0 + hi * 8];
      aF[mt].u[1] = *(const uint4*)&sA[r0 + 16 + hi * 8];
    }
    #pragma unroll
    for (int nt = 0; nt < 4; nt++) {
      int c0 = (wn * 64 + nt * 16 + lm) * 32;
      bF[nt].u[0] = *(const uint4*)&sB[c0 + hi * 8];
      bF[nt].u[1] = *(const uint4*)&sB[c0 + 16 + hi * 8];
    }
    #pragma unroll
    for (int mt = 0; mt < 2; mt++)
      #pragma unroll
      for (int nt = 0; nt < 4; nt++)
        acc[mt][nt] = WMMA_F16(aF[mt].h, bF[nt].h, acc[mt][nt]);
    __syncthreads();
  }

  // epilogue
  #pragma unroll
  for (int mt = 0; mt < 2; mt++) {
    #pragma unroll
    for (int nt = 0; nt < 4; nt++) {
      const int r0 = rowBase + wm * 32 + mt * 16 + hi * 8;
      const int c  = colBase + wn * 64 + nt * 16 + lm;
      const float bv = HAS_BIAS ? bias[c] : 0.0f;
      #pragma unroll
      for (int i = 0; i < 8; i++) {
        float v = acc[mt][nt][i] * scale;
        if (HAS_BIAS) v += bv;
        if (GELU_) v = 0.5f * v * (1.0f + erff(v * 0.70710678118f));
        size_t idx = (size_t)(r0 + i) * (size_t)N + (size_t)c;
        if (RES) v += res[idx];
        if (OUT16) outH[idx] = (_Float16)v; else outF[idx] = v;
      }
    }
  }
}

// ---------------------------------------------------------------------------
// Flash attention (causal): grid (C/128, B*H), 256 threads = 8 waves,
// each wave owns 16 query rows; K/V tiles of 64 staged in LDS per step.
// K tile staged async; V tile transposed through VGPRs. Softmax scale 1/sqrt(D)
// already folded into Q. Mask applied only on diagonal tiles (wave-uniform
// scalar branch) to avoid per-element cndmask VALU on interior tiles.
// ---------------------------------------------------------------------------
#define CDIM 2048
__global__ __launch_bounds__(256) void attn_kernel(const _Float16* __restrict__ Qb,
                                                   const _Float16* __restrict__ Kb,
                                                   const _Float16* __restrict__ Vb,
                                                   _Float16* __restrict__ Ob) {
  __shared__ _Float16 sK[64 * 64];      // [kv][d]
  __shared__ _Float16 sVt[64 * 64];     // [d][kv]
  __shared__ _Float16 sP[8 * 16 * 64];  // per-wave P scratch [m][kv]

  const int t = threadIdx.x, lane = t & 31, w = t >> 5;
  const int lm = lane & 15, hi = lane >> 4;
  const int bh = blockIdx.y;
  const int b = bh >> 4, h = bh & 15;
  const int qbase = blockIdx.x * 128;

  // Q fragments (persistent): rows qbase + w*16 + lm, head slice h*64
  const _Float16* qp = Qb + ((size_t)(b * CDIM + qbase + w * 16 + lm)) * 1024 + h * 64;
  Frag q0, q1;
  q0.u[0] = *(const uint4*)(qp + hi * 8);
  q0.u[1] = *(const uint4*)(qp + 16 + hi * 8);
  q1.u[0] = *(const uint4*)(qp + 32 + hi * 8);
  q1.u[1] = *(const uint4*)(qp + 48 + hi * 8);

  v8f O[4];
  #pragma unroll
  for (int nt = 0; nt < 4; nt++) O[nt] = (v8f){};
  float mrow[8], lrow[8];
  #pragma unroll
  for (int i = 0; i < 8; i++) { mrow[i] = -1e30f; lrow[i] = 0.0f; }

  const int lr = t >> 2, lc = (t & 3) * 16;
  const int wave_qmin = qbase + w * 16;
  const int wave_qmax = wave_qmin + 15;
  const int kend = qbase + 128;
  _Float16* myP = sP + w * 16 * 64;
  const unsigned sK_dst = lds_addr32(&sK[lr * 64 + lc]);

  for (int kb = 0; kb < kend; kb += 64) {
    // stage K tile: async global -> LDS
    const _Float16* kpg = Kb + ((size_t)(b * CDIM + kb + lr)) * 1024 + h * 64 + lc;
    async_b128(sK_dst,      kpg);
    async_b128(sK_dst + 16, kpg + 8);
    // stage V tile transposed (through VGPRs)
    const _Float16* vpg = Vb + ((size_t)(b * CDIM + kb + lr)) * 1024 + h * 64 + lc;
    H8 v0; v0.u = *(const uint4*)(vpg);
    H8 v1; v1.u = *(const uint4*)(vpg + 8);
    #pragma unroll
    for (int j = 0; j < 8; j++) {
      sVt[(lc + j) * 64 + lr]     = v0.e[j];
      sVt[(lc + 8 + j) * 64 + lr] = v1.e[j];
    }
    wait_async0();
    __syncthreads();

    if (kb <= wave_qmax) {
      // S = Q @ K^T for 4 kv-subtiles
      v8f S[4];
      #pragma unroll
      for (int nt = 0; nt < 4; nt++) {
        const int col = (nt * 16 + lm) * 64;
        Frag kf0, kf1;
        kf0.u[0] = *(const uint4*)&sK[col + hi * 8];
        kf0.u[1] = *(const uint4*)&sK[col + 16 + hi * 8];
        kf1.u[0] = *(const uint4*)&sK[col + 32 + hi * 8];
        kf1.u[1] = *(const uint4*)&sK[col + 48 + hi * 8];
        v8f c = (v8f){};
        c = WMMA_F16(q0.h, kf0.h, c);
        c = WMMA_F16(q1.h, kf1.h, c);
        S[nt] = c;
      }
      // causal mask: only diagonal tiles need it (wave-uniform branch)
      if (kb + 63 >= wave_qmin) {
        #pragma unroll
        for (int nt = 0; nt < 4; nt++) {
          const int kv = kb + nt * 16 + lm;
          #pragma unroll
          for (int i = 0; i < 8; i++) {
            const int q = wave_qmin + hi * 8 + i;
            if (kv > q) S[nt][i] = -1e30f;
          }
        }
      }
      // online softmax per row
      #pragma unroll
      for (int i = 0; i < 8; i++) {
        float rmax = fmaxf(fmaxf(S[0][i], S[1][i]), fmaxf(S[2][i], S[3][i]));
        #pragma unroll
        for (int off = 8; off; off >>= 1) rmax = fmaxf(rmax, __shfl_xor(rmax, off, 32));
        const float mnew  = fmaxf(mrow[i], rmax);
        const float alpha = __expf(mrow[i] - mnew);
        mrow[i] = mnew;
        float ps = 0.0f;
        #pragma unroll
        for (int nt = 0; nt < 4; nt++) {
          const float p = __expf(S[nt][i] - mnew);
          ps += p;
          myP[(hi * 8 + i) * 64 + nt * 16 + lm] = (_Float16)p;
        }
        #pragma unroll
        for (int off = 8; off; off >>= 1) ps += __shfl_xor(ps, off, 32);
        lrow[i] = lrow[i] * alpha + ps;
        #pragma unroll
        for (int nt = 0; nt < 4; nt++) O[nt][i] *= alpha;
      }
      // same-wave LDS store->load ordering for P scratch
      asm volatile("s_wait_dscnt 0" ::: "memory");
      // O += P @ V
      #pragma unroll
      for (int kp2 = 0; kp2 < 2; kp2++) {
        Frag pf;
        pf.u[0] = *(const uint4*)&myP[lm * 64 + kp2 * 32 + hi * 8];
        pf.u[1] = *(const uint4*)&myP[lm * 64 + kp2 * 32 + 16 + hi * 8];
        #pragma unroll
        for (int nt = 0; nt < 4; nt++) {
          const int drow = (nt * 16 + lm) * 64;
          Frag vf;
          vf.u[0] = *(const uint4*)&sVt[drow + kp2 * 32 + hi * 8];
          vf.u[1] = *(const uint4*)&sVt[drow + kp2 * 32 + 16 + hi * 8];
          O[nt] = WMMA_F16(pf.h, vf.h, O[nt]);
        }
      }
    }
    __syncthreads();
  }

  // normalize + write
  #pragma unroll
  for (int nt = 0; nt < 4; nt++) {
    #pragma unroll
    for (int i = 0; i < 8; i++) {
      const int q = qbase + w * 16 + hi * 8 + i;
      size_t idx = ((size_t)(b * CDIM + q)) * 1024 + h * 64 + nt * 16 + lm;
      Ob[idx] = (_Float16)(O[nt][i] / lrow[i]);
    }
  }
}

// ---------------------------------------------------------------------------
// Host-side orchestration
// ---------------------------------------------------------------------------
extern "C" void kernel_launch(void* const* d_in, const int* in_sizes, int n_in,
                              void* d_out, int out_size, void* d_ws, size_t ws_size,
                              hipStream_t stream) {
  const float* x     = (const float*)d_in[0];
  const float* ln1_g = (const float*)d_in[1];
  const float* ln1_b = (const float*)d_in[2];
  const float* Wq    = (const float*)d_in[3];
  const float* Wk    = (const float*)d_in[4];
  const float* Wv    = (const float*)d_in[5];
  const float* Wo    = (const float*)d_in[6];
  const float* ln2_g = (const float*)d_in[7];
  const float* ln2_b = (const float*)d_in[8];
  const float* W1    = (const float*)d_in[9];
  const float* b1    = (const float*)d_in[10];
  const float* W2    = (const float*)d_in[11];
  const float* b2    = (const float*)d_in[12];
  float* out = (float*)d_out;

  const size_t MB = 1ull << 20;
  char* ws = (char*)d_ws;
  _Float16* h16   = (_Float16*)(ws + 0);        // 16 MB
  _Float16* Qb    = (_Float16*)(ws + 16 * MB);  // 16 MB
  _Float16* Kb    = (_Float16*)(ws + 32 * MB);  // 16 MB
  _Float16* Vb    = (_Float16*)(ws + 48 * MB);  // 16 MB
  _Float16* attnO = (_Float16*)(ws + 64 * MB);  // 16 MB
  float*    x1    = (float*)   (ws + 80 * MB);  // 32 MB
  _Float16* h2    = (_Float16*)(ws + 112 * MB); // 16 MB
  _Float16* gbuf  = (_Float16*)(ws + 0);        // 64 MB, aliases h16/Q/K/V (dead by then)

  const int M = 4 * 2048;  // 8192 rows
  const dim3 blk(256);

  // LN1
  ln_kernel<<<dim3(M), blk, 0, stream>>>(x, ln1_g, ln1_b, h16);
  // QKV projections (Q gets 1/sqrt(64) = 0.125 folded in)
  gemm_f16<false, false, false, true><<<dim3(8, 64), blk, 0, stream>>>(
      h16, Wq, nullptr, nullptr, nullptr, Qb, M, 1024, 1024, 0.125f);
  gemm_f16<false, false, false, true><<<dim3(8, 64), blk, 0, stream>>>(
      h16, Wk, nullptr, nullptr, nullptr, Kb, M, 1024, 1024, 1.0f);
  gemm_f16<false, false, false, true><<<dim3(8, 64), blk, 0, stream>>>(
      h16, Wv, nullptr, nullptr, nullptr, Vb, M, 1024, 1024, 1.0f);
  // causal flash attention
  attn_kernel<<<dim3(16, 64), blk, 0, stream>>>(Qb, Kb, Vb, attnO);
  // output projection + residual -> x1 (f32)
  gemm_f16<false, false, true, false><<<dim3(8, 64), blk, 0, stream>>>(
      attnO, Wo, nullptr, x, x1, nullptr, M, 1024, 1024, 1.0f);
  // LN2
  ln_kernel<<<dim3(M), blk, 0, stream>>>(x1, ln2_g, ln2_b, h2);
  // FFN1 + bias + exact GELU -> g (f16)
  gemm_f16<true, true, false, true><<<dim3(32, 64), blk, 0, stream>>>(
      h2, W1, b1, nullptr, nullptr, gbuf, M, 4096, 1024, 1.0f);
  // FFN2 + bias + residual -> out (f32)
  gemm_f16<true, false, true, false><<<dim3(8, 64), blk, 0, stream>>>(
      gbuf, W2, b2, x1, out, nullptr, M, 1024, 4096, 1.0f);
}
